// Block_37890201485646
// MI455X (gfx1250) — matrix operand order repaired and checked by reading, not compiled, not run on hardware
//
#include <hip/hip_runtime.h>

typedef __attribute__((ext_vector_type(16))) __bf16 v16bf;
typedef __attribute__((ext_vector_type(8)))  float  v8f;

#define BT   8192    // B*T rows
#define DM   1024
#define DFFN 4096
#define NH   16
#define HDIM 64
#define SEQ  2048
#define NBAT 4

// Build a 16-element bf16 fragment from two 16-byte chunks (LDS or global).
__device__ __forceinline__ v16bf frag2(const void* p0, const void* p1) {
    union { uint4 u[2]; v16bf v; } t;
    t.u[0] = *reinterpret_cast<const uint4*>(p0);
    t.u[1] = *reinterpret_cast<const uint4*>(p1);
    return t.v;
}

// ---------------------------------------------------------------------------
// One-time weight preparation: fp32 -> bf16 with transpose to [N, K] so that
// WMMA B-fragments are K-contiguous (two b128 loads per fragment).
// ---------------------------------------------------------------------------
// in: [K, N] fp32 row-major -> out: [N, K] bf16
__global__ __launch_bounds__(256)
void cvt_w_T(const float* __restrict__ in, __bf16* __restrict__ out, int K, int N) {
    __shared__ float t[32][33];
    const int n0 = blockIdx.x * 32, k0 = blockIdx.y * 32;
    const int tx = threadIdx.x, ty = threadIdx.y;     // 32 x 8
#pragma unroll
    for (int j = 0; j < 4; ++j)
        t[ty + j * 8][tx] = in[(size_t)(k0 + ty + j * 8) * N + n0 + tx];
    __syncthreads();
#pragma unroll
    for (int j = 0; j < 4; ++j)
        out[(size_t)(n0 + ty + j * 8) * K + k0 + tx] = (__bf16)t[tx][ty + j * 8];
}

// Wq/Wk/Wv: [H, D, HD] fp32 -> out [H*HD, D] bf16 (n = h*HD+e, k = d)
__global__ __launch_bounds__(256)
void cvt_headw_T(const float* __restrict__ in, __bf16* __restrict__ out) {
    __shared__ float t[32][33];
    const int h = blockIdx.z;
    const int d0 = blockIdx.x * 32, e0 = blockIdx.y * 32;
    const int tx = threadIdx.x, ty = threadIdx.y;
    const float* src = in + (size_t)h * DM * HDIM;
#pragma unroll
    for (int j = 0; j < 4; ++j)
        t[ty + j * 8][tx] = src[(size_t)(d0 + ty + j * 8) * HDIM + e0 + tx];
    __syncthreads();
#pragma unroll
    for (int j = 0; j < 4; ++j)
        out[(size_t)(h * HDIM + e0 + ty + j * 8) * DM + d0 + tx] = (__bf16)t[tx][ty + j * 8];
}

// V: [B*T, D] bf16 -> vT [B*H, HD, T] bf16 so P*V B-fragments are contiguous.
__global__ __launch_bounds__(256)
void transpose_v(const __bf16* __restrict__ in, __bf16* __restrict__ out) {
    __shared__ __bf16 t[32][34];
    const int bh = blockIdx.z;
    const int bat = bh / NH, h = bh % NH;
    const int t0 = blockIdx.x * 32, e0 = blockIdx.y * 32;
    const int tx = threadIdx.x, ty = threadIdx.y;
#pragma unroll
    for (int j = 0; j < 4; ++j)
        t[ty + j * 8][tx] = in[(size_t)(bat * SEQ + t0 + ty + j * 8) * DM + h * HDIM + e0 + tx];
    __syncthreads();
#pragma unroll
    for (int j = 0; j < 4; ++j)
        out[((size_t)bh * HDIM + e0 + ty + j * 8) * SEQ + t0 + tx] = t[tx][ty + j * 8];
}

// ---------------------------------------------------------------------------
// LayerNorm: one block (256 thr) per row of 1024, vectorized, bf16 output.
// ---------------------------------------------------------------------------
__global__ __launch_bounds__(256)
void ln_bf16_kernel(const float* __restrict__ x, const float* __restrict__ g,
                    const float* __restrict__ be, __bf16* __restrict__ out) {
    __shared__ float red[16];
    const int row = blockIdx.x;
    const int tid = threadIdx.x;
    const float4 f = reinterpret_cast<const float4*>(x + (size_t)row * DM)[tid];
    float s  = f.x + f.y + f.z + f.w;
    float s2 = f.x * f.x + f.y * f.y + f.z * f.z + f.w * f.w;
#pragma unroll
    for (int m = 16; m >= 1; m >>= 1) {
        s  += __shfl_xor(s,  m, 32);
        s2 += __shfl_xor(s2, m, 32);
    }
    if ((tid & 31) == 0) { red[tid >> 5] = s; red[(tid >> 5) + 8] = s2; }
    __syncthreads();
    s = 0.f; s2 = 0.f;
#pragma unroll
    for (int i = 0; i < 8; ++i) { s += red[i]; s2 += red[i + 8]; }
    const float mu   = s * (1.0f / DM);
    const float var  = s2 * (1.0f / DM) - mu * mu;
    const float rstd = rsqrtf(var + 1e-5f);
    const float4 gv  = reinterpret_cast<const float4*>(g)[tid];
    const float4 bv  = reinterpret_cast<const float4*>(be)[tid];
    union { __bf16 h[4]; uint2 u; } o;
    o.h[0] = (__bf16)((f.x - mu) * rstd * gv.x + bv.x);
    o.h[1] = (__bf16)((f.y - mu) * rstd * gv.y + bv.y);
    o.h[2] = (__bf16)((f.z - mu) * rstd * gv.z + bv.z);
    o.h[3] = (__bf16)((f.w - mu) * rstd * gv.w + bv.w);
    *reinterpret_cast<uint2*>(&out[(size_t)row * DM + tid * 4]) = o.u;
}

// ---------------------------------------------------------------------------
// GEMM: C[M,N] = A[M,K](bf16) * Bt[N,K](bf16 pre-transposed), compile-time
// fused epilogue. Block tile 128x64, K-step 32, double-buffered LDS, 8 waves;
// each wave computes 32x32 (4 wmma / K-step).
// ---------------------------------------------------------------------------
template <bool BIAS, bool RESID, bool RELU, bool WF>
__global__ __launch_bounds__(256)
void gemm_bf16_128x64(const __bf16* __restrict__ A, const __bf16* __restrict__ Bt,
                      const float* __restrict__ bias, const float* __restrict__ resid,
                      float* __restrict__ outF, __bf16* __restrict__ outB,
                      int M, int N, int K) {
    __shared__ __bf16 As[2][128][40];
    __shared__ __bf16 Bs[2][64][40];
    const int tid  = threadIdx.x;
    const int lane = tid & 31;
    const int w    = tid >> 5;
    const int bM   = blockIdx.y * 128;
    const int bN   = blockIdx.x * 64;
    const int wr   = (w & 3) * 32;
    const int wc   = (w >> 2) * 32;
    const int m    = lane & 15;
    const int kh   = (lane >> 4) * 8;
    const int kh16 = (lane >> 4) * 16;

    // staging assignments (one b128 chunk each)
    const int ra0 = tid >> 2,           oa0 = (tid & 3) * 8;
    const int ra1 = (tid + 256) >> 2,   oa1 = (tid & 3) * 8;
    const int rb  = tid >> 2,           ob  = (tid & 3) * 8;
    const __bf16* gA0 = A  + (size_t)(bM + ra0) * K + oa0;
    const __bf16* gA1 = A  + (size_t)(bM + ra1) * K + oa1;
    const __bf16* gB  = Bt + (size_t)(bN + rb)  * K + ob;

    v8f acc[2][2] = {{{}, {}}, {{}, {}}};

    // preload tile 0
    uint4 cA0 = *reinterpret_cast<const uint4*>(gA0);
    uint4 cA1 = *reinterpret_cast<const uint4*>(gA1);
    uint4 cB  = *reinterpret_cast<const uint4*>(gB);
    *reinterpret_cast<uint4*>(&As[0][ra0][oa0]) = cA0;
    *reinterpret_cast<uint4*>(&As[0][ra1][oa1]) = cA1;
    *reinterpret_cast<uint4*>(&Bs[0][rb][ob])   = cB;

    int p = 0;
    for (int k0 = 0; k0 < K; k0 += 32) {
        __syncthreads();
        const bool more = (k0 + 32) < K;
        uint4 nA0, nA1, nB;
        if (more) {       // overlap next tile's global loads with WMMA
            nA0 = *reinterpret_cast<const uint4*>(gA0 + k0 + 32);
            nA1 = *reinterpret_cast<const uint4*>(gA1 + k0 + 32);
            nB  = *reinterpret_cast<const uint4*>(gB  + k0 + 32);
        }

        v16bf a[2], b[2];
#pragma unroll
        for (int r = 0; r < 2; ++r)
            a[r] = frag2(&As[p][wr + r * 16 + m][kh], &As[p][wr + r * 16 + m][16 + kh]);
#pragma unroll
        for (int c = 0; c < 2; ++c)
            b[c] = frag2(&Bs[p][wc + c * 16 + m][kh], &Bs[p][wc + c * 16 + m][16 + kh]);
#pragma unroll
        for (int r = 0; r < 2; ++r)
#pragma unroll
            for (int c = 0; c < 2; ++c)
                acc[r][c] = __builtin_amdgcn_wmma_f32_16x16x32_bf16(
                    false, a[r], false, b[c], (short)0, acc[r][c], false, false);

        if (more) {
            *reinterpret_cast<uint4*>(&As[p ^ 1][ra0][oa0]) = nA0;
            *reinterpret_cast<uint4*>(&As[p ^ 1][ra1][oa1]) = nA1;
            *reinterpret_cast<uint4*>(&Bs[p ^ 1][rb][ob])   = nB;
            p ^= 1;
        }
    }

    const int n = lane & 15, half = lane >> 4;
#pragma unroll
    for (int r = 0; r < 2; ++r) {
#pragma unroll
        for (int c = 0; c < 2; ++c) {
            const int row0 = bM + wr + r * 16 + 8 * half;
            const int col  = bN + wc + c * 16 + n;
            float bc = 0.0f;
            if (BIAS) bc = bias[col];
            size_t idx = (size_t)row0 * N + col;
#pragma unroll
            for (int i = 0; i < 8; ++i) {
                float v = acc[r][c][i];
                if (BIAS)  v += bc;
                if (RESID) v += resid[idx];
                if (RELU)  v = fmaxf(v, 0.0f);
                if (WF) outF[idx] = v;
                else    outB[idx] = (__bf16)v;
                idx += N;
            }
        }
    }
}

// ---------------------------------------------------------------------------
// Flash-attention: 128 query rows per block per (b,h). Key blocks of 64,
// double-buffered K/V tiles, online softmax, causal.
// Q/K as [B*T, H*HD] bf16; V pre-transposed [B*H, HD, T].
// ---------------------------------------------------------------------------
__global__ __launch_bounds__(256)
void attn_kernel(const __bf16* __restrict__ Q, const __bf16* __restrict__ Kp,
                 const __bf16* __restrict__ Vt, __bf16* __restrict__ ctx) {
    __shared__ __bf16 Qs[128][72];
    __shared__ __bf16 Ks[2][64][72];
    __shared__ __bf16 Vs[2][64][72];   // [e][token] (transposed V tile)
    __shared__ __bf16 Ps[8][16][72];

    const int tid   = threadIdx.x;
    const int lane  = tid & 31;
    const int w     = tid >> 5;
    const int qbase = blockIdx.x * 128;
    const int bh    = blockIdx.y;
    const int bat   = bh / NH, h = bh % NH;
    const size_t rowbase = (size_t)bat * SEQ;
    const int hc = h * HDIM;

    for (int c = tid; c < 1024; c += 256) {
        int r = c >> 3, off = (c & 7) * 8;
        *reinterpret_cast<uint4*>(&Qs[r][off]) =
            *reinterpret_cast<const uint4*>(&Q[(rowbase + qbase + r) * DM + hc + off]);
    }

    // staging: two b128 chunks each for K and V per thread
    const int r0 = tid >> 3,          o0 = (tid & 7) * 8;
    const int r1 = (tid + 256) >> 3,  o1 = (tid & 7) * 8;
    const __bf16* gK0 = Kp + (rowbase + r0) * DM + hc + o0;   // +64*DM per kb
    const __bf16* gK1 = Kp + (rowbase + r1) * DM + hc + o1;
    const __bf16* gV0 = Vt + ((size_t)bh * HDIM + r0) * SEQ + o0;  // +64 per kb
    const __bf16* gV1 = Vt + ((size_t)bh * HDIM + r1) * SEQ + o1;

    float mi[8], li[8];
    v8f O[4];
#pragma unroll
    for (int i = 0; i < 8; ++i) { mi[i] = -3.0e38f; li[i] = 0.0f; }
#pragma unroll
    for (int t = 0; t < 4; ++t) O[t] = (v8f){};

    const int n = lane & 15, half = lane >> 4;
    const int kh = half * 8, kh16 = half * 16;
    const int trow = qbase + w * 16 + 8 * half;   // +i -> lane's token row

    // preload key-block 0
    *reinterpret_cast<uint4*>(&Ks[0][r0][o0]) = *reinterpret_cast<const uint4*>(gK0);
    *reinterpret_cast<uint4*>(&Ks[0][r1][o1]) = *reinterpret_cast<const uint4*>(gK1);
    *reinterpret_cast<uint4*>(&Vs[0][r0][o0]) = *reinterpret_cast<const uint4*>(gV0);
    *reinterpret_cast<uint4*>(&Vs[0][r1][o1]) = *reinterpret_cast<const uint4*>(gV1);

    const int nkb = qbase / 64 + 2;               // causal: only needed key blocks
    int p = 0;
    for (int kb0 = 0; kb0 < nkb; ++kb0) {
        __syncthreads();
        const bool more = (kb0 + 1) < nkb;
        uint4 nK0, nK1, nV0, nV1;
        if (more) {
            const size_t dk = (size_t)(kb0 + 1) * 64 * DM;
            const size_t dv = (size_t)(kb0 + 1) * 64;
            nK0 = *reinterpret_cast<const uint4*>(gK0 + dk);
            nK1 = *reinterpret_cast<const uint4*>(gK1 + dk);
            nV0 = *reinterpret_cast<const uint4*>(gV0 + dv);
            nV1 = *reinterpret_cast<const uint4*>(gV1 + dv);
        }

        // S = Q * K^T  (16x64 strip per wave); all fragments are b128 loads
        v8f S[4];
#pragma unroll
        for (int t = 0; t < 4; ++t) S[t] = (v8f){};
#pragma unroll
        for (int ks = 0; ks < 2; ++ks) {
            v16bf a = frag2(&Qs[w * 16 + n][ks * 32 + kh], &Qs[w * 16 + n][ks * 32 + 16 + kh]);
#pragma unroll
            for (int t = 0; t < 4; ++t) {
                v16bf bkt = frag2(&Ks[p][t * 16 + n][ks * 32 + kh],
                                  &Ks[p][t * 16 + n][ks * 32 + 16 + kh]);
                S[t] = __builtin_amdgcn_wmma_f32_16x16x32_bf16(
                    false, a, false, bkt, (short)0, S[t], false, false);
            }
        }

        // causal mask + online softmax (row reductions over 16-lane halves)
#pragma unroll
        for (int i = 0; i < 8; ++i) {
            int tok = trow + i;
            float mx = -3.0e38f;
#pragma unroll
            for (int t = 0; t < 4; ++t) {
                int col = kb0 * 64 + t * 16 + n;
                float s = S[t][i] * 0.125f;     // 1/sqrt(HD)
                if (col > tok) s = -3.0e38f;
                S[t][i] = s;
                mx = fmaxf(mx, s);
            }
#pragma unroll
            for (int msk = 8; msk >= 1; msk >>= 1)
                mx = fmaxf(mx, __shfl_xor(mx, msk, 32));
            float nm = fmaxf(mi[i], mx);
            float sum = 0.0f;
#pragma unroll
            for (int t = 0; t < 4; ++t) {
                float pr = __expf(S[t][i] - nm);
                S[t][i] = pr;
                sum += pr;
            }
#pragma unroll
            for (int msk = 8; msk >= 1; msk >>= 1)
                sum += __shfl_xor(sum, msk, 32);
            float sc = __expf(mi[i] - nm);
            li[i] = li[i] * sc + sum;
            mi[i] = nm;
#pragma unroll
            for (int t = 0; t < 4; ++t) O[t][i] *= sc;
        }

        // stage P (C-layout) through LDS into A-layout for P*V
#pragma unroll
        for (int i = 0; i < 8; ++i) {
            int r = i + 8 * half;
#pragma unroll
            for (int t = 0; t < 4; ++t)
                Ps[w][r][t * 16 + n] = (__bf16)S[t][i];
        }
#pragma unroll
        for (int ks = 0; ks < 2; ++ks) {
            v16bf a = frag2(&Ps[w][n][ks * 32 + kh], &Ps[w][n][ks * 32 + 16 + kh]);
#pragma unroll
            for (int t = 0; t < 4; ++t) {
                v16bf bv = frag2(&Vs[p][t * 16 + n][ks * 32 + kh],
                                 &Vs[p][t * 16 + n][ks * 32 + 16 + kh]);
                O[t] = __builtin_amdgcn_wmma_f32_16x16x32_bf16(
                    false, a, false, bv, (short)0, O[t], false, false);
            }
        }

        if (more) {
            *reinterpret_cast<uint4*>(&Ks[p ^ 1][r0][o0]) = nK0;
            *reinterpret_cast<uint4*>(&Ks[p ^ 1][r1][o1]) = nK1;
            *reinterpret_cast<uint4*>(&Vs[p ^ 1][r0][o0]) = nV0;
            *reinterpret_cast<uint4*>(&Vs[p ^ 1][r1][o1]) = nV1;
            p ^= 1;
        }
    }

    // normalize and write concat-heads ctx [B*T, D] bf16
#pragma unroll
    for (int i = 0; i < 8; ++i) {
        float inv = 1.0f / li[i];
        int tok = trow + i;
#pragma unroll
        for (int t = 0; t < 4; ++t)
            ctx[(rowbase + tok) * DM + hc + t * 16 + n] = (__bf16)(O[t][i] * inv);
    }
}

// ---------------------------------------------------------------------------
// Host-side orchestration
// ---------------------------------------------------------------------------
extern "C" void kernel_launch(void* const* d_in, const int* in_sizes, int n_in,
                              void* d_out, int out_size, void* d_ws, size_t ws_size,
                              hipStream_t stream) {
    const float* x   = (const float*)d_in[0];
    const float* Wq  = (const float*)d_in[1];
    const float* Wk  = (const float*)d_in[2];
    const float* Wv  = (const float*)d_in[3];
    const float* Wo  = (const float*)d_in[4];
    const float* bo  = (const float*)d_in[5];
    const float* W1  = (const float*)d_in[6];
    const float* b1  = (const float*)d_in[7];
    const float* W2  = (const float*)d_in[8];
    const float* b2  = (const float*)d_in[9];
    const float* g1  = (const float*)d_in[10];
    const float* be1 = (const float*)d_in[11];
    const float* g2  = (const float*)d_in[12];
    const float* be2 = (const float*)d_in[13];
    float* out = (float*)d_out;

    char* ws = (char*)d_ws;
    size_t off = 0;
    auto alloc = [&](size_t bytes) -> void* {
        void* p = ws + off;
        off = (off + bytes + 255) & ~(size_t)255;
        return p;
    };
    __bf16* wqb  = (__bf16*)alloc((size_t)DM * DM * 2);      // [N=D, K=D]
    __bf16* wkb  = (__bf16*)alloc((size_t)DM * DM * 2);
    __bf16* wvb  = (__bf16*)alloc((size_t)DM * DM * 2);
    __bf16* wob  = (__bf16*)alloc((size_t)DM * DM * 2);
    __bf16* w1b  = (__bf16*)alloc((size_t)DM * DFFN * 2);    // [N=DFF, K=D]
    __bf16* w2b  = (__bf16*)alloc((size_t)DFFN * DM * 2);    // [N=D, K=DFF]
    __bf16* hb   = (__bf16*)alloc((size_t)BT * DM * 2);
    __bf16* qb   = (__bf16*)alloc((size_t)BT * DM * 2);
    __bf16* kbuf = (__bf16*)alloc((size_t)BT * DM * 2);
    __bf16* vbuf = (__bf16*)alloc((size_t)BT * DM * 2);
    __bf16* vtb  = (__bf16*)alloc((size_t)BT * DM * 2);      // [B*H, HD, T]
    __bf16* ctxb = (__bf16*)alloc((size_t)BT * DM * 2);
    float*  x1f  = (float*)alloc((size_t)BT * DM * 4);
    __bf16* h2b  = (__bf16*)alloc((size_t)BT * DM * 2);
    __bf16* ffb  = (__bf16*)alloc((size_t)BT * DFFN * 2);

    dim3 blk(256);
    dim3 tblk(32, 8);

    // weights -> bf16 transposed to [N, K]
    dim3 ghw(DM / 32, HDIM / 32, NH);
    cvt_headw_T<<<ghw, tblk, 0, stream>>>(Wq, wqb);
    cvt_headw_T<<<ghw, tblk, 0, stream>>>(Wk, wkb);
    cvt_headw_T<<<ghw, tblk, 0, stream>>>(Wv, wvb);
    cvt_w_T<<<dim3(DM / 32, DM / 32), tblk, 0, stream>>>(Wo, wob, DM, DM);
    cvt_w_T<<<dim3(DFFN / 32, DM / 32), tblk, 0, stream>>>(W1, w1b, DM, DFFN);
    cvt_w_T<<<dim3(DM / 32, DFFN / 32), tblk, 0, stream>>>(W2, w2b, DFFN, DM);

    // h = LN(x)
    ln_bf16_kernel<<<BT, blk, 0, stream>>>(x, g1, be1, hb);

    // QKV projections (bf16 out only)
    dim3 gqkv(DM / 64, BT / 128);
    gemm_bf16_128x64<false, false, false, false><<<gqkv, blk, 0, stream>>>(
        hb, wqb, nullptr, nullptr, nullptr, qb, BT, DM, DM);
    gemm_bf16_128x64<false, false, false, false><<<gqkv, blk, 0, stream>>>(
        hb, wkb, nullptr, nullptr, nullptr, kbuf, BT, DM, DM);
    gemm_bf16_128x64<false, false, false, false><<<gqkv, blk, 0, stream>>>(
        hb, wvb, nullptr, nullptr, nullptr, vbuf, BT, DM, DM);

    // V -> [B*H, HD, T] for contiguous P*V fragments
    transpose_v<<<dim3(SEQ / 32, HDIM / 32, NBAT * NH), tblk, 0, stream>>>(vbuf, vtb);

    // causal flash attention -> ctx (concat heads)
    dim3 ga(SEQ / 128, NBAT * NH);
    attn_kernel<<<ga, blk, 0, stream>>>(qb, kbuf, vtb, ctxb);

    // x1 = x + ctx*Wo + bo  (bias + residual, fp32 out)
    gemm_bf16_128x64<true, true, false, true><<<gqkv, blk, 0, stream>>>(
        ctxb, wob, bo, x, x1f, nullptr, BT, DM, DM);

    // h2 = LN(x1)
    ln_bf16_kernel<<<BT, blk, 0, stream>>>(x1f, g2, be2, h2b);

    // ff = relu(h2*W1 + b1)  (bias + relu, bf16 out)
    dim3 gff(DFFN / 64, BT / 128);
    gemm_bf16_128x64<true, false, true, false><<<gff, blk, 0, stream>>>(
        h2b, w1b, b1, nullptr, nullptr, ffb, BT, DFFN, DM);

    // out = x1 + ff*W2 + b2  (bias + residual, fp32 out)
    gemm_bf16_128x64<true, true, false, true><<<gqkv, blk, 0, stream>>>(
        ffb, w2b, b2, x1f, out, nullptr, BT, DM, DFFN);
}
